// TrajectoryGeneratorAR_goal_76802605187713
// MI455X (gfx1250) — matrix operand order repaired and checked by reading, not compiled, not run on hardware
//
#include <hip/hip_runtime.h>
#include <hip/hip_bf16.h>

#define NB   16384   // B = S*N
#define NS   256     // samples
#define NA   64      // agents per sample
#define PLEN 12

typedef __attribute__((ext_vector_type(16))) _Float16 v16h;
typedef __attribute__((ext_vector_type(8)))  _Float16 v8h;
typedef __attribute__((ext_vector_type(8)))  float    v8f;

__device__ __forceinline__ float sigmf(float x) { return 1.0f / (1.0f + __expf(-x)); }
__device__ __forceinline__ float reluf(float x) { return fmaxf(x, 0.0f); }

// ---------------------------------------------------------------------------
// Phase 1: history encoder (8-step LSTM, H=16) + state init
// ---------------------------------------------------------------------------
__global__ __launch_bounds__(256) void enc_kernel(
    const float* __restrict__ traj_rel,  // (8,B,2)
    const float* __restrict__ obs_pos,   // (8,B,2)
    const float* __restrict__ goal,      // (B,2)
    const float* __restrict__ We, const float* __restrict__ be,     // (2,16),(16)
    const float* __restrict__ Wih, const float* __restrict__ Whh,   // (16,64)
    const float* __restrict__ bl,                                   // (64)
    float* __restrict__ hbuf, float* __restrict__ cbuf, float* __restrict__ ctxb,
    float* __restrict__ out2, float* __restrict__ curr, float* __restrict__ relg)
{
    __shared__ float sWe[32], sbe[16], sWih[1024], sWhh[1024], sbl[64];
    int tid = threadIdx.x;
    for (int t = tid; t < 1024; t += 256) { sWih[t] = Wih[t]; sWhh[t] = Whh[t]; }
    if (tid < 32) sWe[tid] = We[tid];
    if (tid < 16) sbe[tid] = be[tid];
    if (tid < 64) sbl[tid] = bl[tid];
    __syncthreads();

    int b = blockIdx.x * 256 + tid;
    float h[16], c[16], hn[16], x[16];
#pragma unroll
    for (int k = 0; k < 16; k++) { h[k] = 0.f; c[k] = 0.f; }

    for (int t = 0; t < 8; t++) {
        float x0 = traj_rel[((size_t)t * NB + b) * 2 + 0];
        float x1 = traj_rel[((size_t)t * NB + b) * 2 + 1];
#pragma unroll
        for (int k = 0; k < 16; k++) x[k] = reluf(x0 * sWe[k] + x1 * sWe[16 + k] + sbe[k]);
#pragma unroll
        for (int k = 0; k < 16; k++) {
            float gi = sbl[k], gf = sbl[16 + k], gg = sbl[32 + k], go = sbl[48 + k];
#pragma unroll
            for (int q = 0; q < 16; q++) {
                float xq = x[q], hq = h[q];
                gi += xq * sWih[q * 64 + k]      + hq * sWhh[q * 64 + k];
                gf += xq * sWih[q * 64 + 16 + k] + hq * sWhh[q * 64 + 16 + k];
                gg += xq * sWih[q * 64 + 32 + k] + hq * sWhh[q * 64 + 32 + k];
                go += xq * sWih[q * 64 + 48 + k] + hq * sWhh[q * 64 + 48 + k];
            }
            float cn = sigmf(gf) * c[k] + sigmf(gi) * tanhf(gg);
            c[k] = cn;
            hn[k] = sigmf(go) * tanhf(cn);
        }
#pragma unroll
        for (int k = 0; k < 16; k++) h[k] = hn[k];
    }
#pragma unroll
    for (int k = 0; k < 16; k++) {
        hbuf[b * 16 + k] = h[k];
        cbuf[b * 16 + k] = 0.f;     // decoder starts with c = 0
        ctxb[b * 16 + k] = 0.f;     // ctx starts at 0
    }
    float p0 = obs_pos[((size_t)7 * NB + b) * 2 + 0];
    float p1 = obs_pos[((size_t)7 * NB + b) * 2 + 1];
    out2[b * 2 + 0] = traj_rel[((size_t)7 * NB + b) * 2 + 0];
    out2[b * 2 + 1] = traj_rel[((size_t)7 * NB + b) * 2 + 1];
    curr[b * 2 + 0] = p0;                   curr[b * 2 + 1] = p1;
    relg[b * 2 + 0] = goal[b * 2 + 0] - p0; relg[b * 2 + 1] = goal[b * 2 + 1] - p1;
}

// ---------------------------------------------------------------------------
// Phase 2a (per step): decoder input embed + LSTM cell; also emit h as f16
// ---------------------------------------------------------------------------
__global__ __launch_bounds__(256) void dec_step(
    const float* __restrict__ ctxb, const float* __restrict__ out2,
    const float* __restrict__ Wi, const float* __restrict__ bi,     // (18,16),(16)
    const float* __restrict__ Wih, const float* __restrict__ Whh,   // (16,64)
    const float* __restrict__ bl,                                   // (64)
    float* __restrict__ hbuf, float* __restrict__ cbuf, _Float16* __restrict__ h16)
{
    __shared__ float sWi[288], sbi[16], sWih[1024], sWhh[1024], sbl[64];
    int tid = threadIdx.x;
    for (int t = tid; t < 1024; t += 256) { sWih[t] = Wih[t]; sWhh[t] = Whh[t]; }
    for (int t = tid; t < 288; t += 256) sWi[t] = Wi[t];
    if (tid < 16) sbi[tid] = bi[tid];
    if (tid < 64) sbl[tid] = bl[tid];
    __syncthreads();

    int b = blockIdx.x * 256 + tid;
    float inp[18], x[16], h[16], c[16], hn[16];
#pragma unroll
    for (int k = 0; k < 16; k++) inp[k] = ctxb[b * 16 + k];
    inp[16] = out2[b * 2 + 0]; inp[17] = out2[b * 2 + 1];
#pragma unroll
    for (int k = 0; k < 16; k++) {
        float a = sbi[k];
#pragma unroll
        for (int q = 0; q < 18; q++) a += inp[q] * sWi[q * 16 + k];
        x[k] = reluf(a);
    }
#pragma unroll
    for (int k = 0; k < 16; k++) { h[k] = hbuf[b * 16 + k]; c[k] = cbuf[b * 16 + k]; }
#pragma unroll
    for (int k = 0; k < 16; k++) {
        float gi = sbl[k], gf = sbl[16 + k], gg = sbl[32 + k], go = sbl[48 + k];
#pragma unroll
        for (int q = 0; q < 16; q++) {
            float xq = x[q], hq = h[q];
            gi += xq * sWih[q * 64 + k]      + hq * sWhh[q * 64 + k];
            gf += xq * sWih[q * 64 + 16 + k] + hq * sWhh[q * 64 + 16 + k];
            gg += xq * sWih[q * 64 + 32 + k] + hq * sWhh[q * 64 + 32 + k];
            go += xq * sWih[q * 64 + 48 + k] + hq * sWhh[q * 64 + 48 + k];
        }
        float cn = sigmf(gf) * c[k] + sigmf(gi) * tanhf(gg);
        c[k] = cn;
        hn[k] = sigmf(go) * tanhf(cn);
    }
#pragma unroll
    for (int k = 0; k < 16; k++) {
        hbuf[b * 16 + k] = hn[k];
        cbuf[b * 16 + k] = c[k];
        h16[b * 16 + k]  = (_Float16)hn[k];
    }
}

// ---------------------------------------------------------------------------
// Phase 2b (per step): pairwise pooling via WMMA.
// One block per sample; 8 waves; wave w owns rows i = 8w..8w+7.
// Layer1 (transposed):  m1^T(64ch x 16 pairs) = W1^T(64x32) @ feat^T(32x16)
//    -> 4x wmma, A = resident W1^T fragments, B = per-tile feat fragment
//    -> lane holds 8 contiguous channels of one pair -> packed b128 staging store
// Layer2 (normal):      out(16 pairs x 16H) = m1(16x64) @ W2(64x16)
//    -> 2x chained wmma, A halves are contiguous b128 loads from staging
// ---------------------------------------------------------------------------
__global__ __launch_bounds__(256) void pool_step(
    const float* __restrict__ curr, const _Float16* __restrict__ h16g,
    const int* __restrict__ nei,
    const float* __restrict__ Wp, const float* __restrict__ bp,   // (2,16),(16)
    const float* __restrict__ W1, const float* __restrict__ b1,   // (32,64),(64)
    const float* __restrict__ W2, const float* __restrict__ b2,   // (64,16),(16)
    float* __restrict__ ctxb)
{
    __shared__ float    s_pos[NA * 2];
    __shared__ _Float16 s_hid[NA * 16];
    __shared__ float    s_mask[NA * NA];
    __shared__ _Float16 s_w1[32 * 64];
    __shared__ _Float16 s_w2[64 * 16];
    __shared__ float    s_b1[64], s_b2[16], s_wp[32], s_bp[16];
    __shared__ _Float16 s_stage[8][16 * 64];   // per-wave staging: [pair][channel]

    const int s = blockIdx.x;
    const int tid = threadIdx.x;
    for (int t = tid; t < NA * 2; t += 256) s_pos[t] = curr[s * NA * 2 + t];
    for (int t = tid; t < NA * 16; t += 256) s_hid[t] = h16g[(size_t)s * NA * 16 + t];
    for (int t = tid; t < NA * NA; t += 256) s_mask[t] = (float)nei[(size_t)s * NA * NA + t];
    for (int t = tid; t < 32 * 64; t += 256) s_w1[t] = (_Float16)W1[t];
    for (int t = tid; t < 64 * 16; t += 256) s_w2[t] = (_Float16)W2[t];
    if (tid < 64) s_b1[tid] = b1[tid];
    if (tid < 16) s_b2[tid] = b2[tid];
    if (tid < 32) s_wp[tid] = Wp[tid];
    if (tid < 16) s_bp[tid] = bp[tid];
    __syncthreads();

    const int wave = tid >> 5, lane = tid & 31;
    const int r = lane & 15, hi = lane >> 4;

    // Resident A fragments = W1^T[nb*16 .. +15, 0..31] (16x32 f16).
    // A layout: lane(r,hi): row M=r (-> ch nb*16+r); elem e -> K=hi*16+e
    v16h aw1[4];
#pragma unroll
    for (int nb = 0; nb < 4; nb++) {
#pragma unroll
        for (int e = 0; e < 16; e++) {
            const int k = hi * 16 + e;
            aw1[nb][e] = s_w1[k * 64 + nb * 16 + r];
        }
    }
    // Resident B fragments = W2 (64x16).  B layout: lane(r,hi): col N=r; e<8 -> K=hi*8+e ; e>=8 -> K=16+hi*8+(e-8)
    v16h bw2[2];
#pragma unroll
    for (int kb = 0; kb < 2; kb++) {
#pragma unroll
        for (int e = 0; e < 16; e++) bw2[kb][e] = s_w2[(kb * 32 + ((e < 8) ? (hi * 8 + e) : (16 + hi * 8 + (e - 8)))) * 16 + r];
    }
    // pe weights for all 16 channels (consumed by hi==0 lanes of the feat fragment)
    float wx[16], wy[16], pb[16];
#pragma unroll
    for (int e = 0; e < 16; e++) { wx[e] = s_wp[e]; wy[e] = s_wp[16 + e]; pb[e] = s_bp[e]; }
    // layer1 bias, per accumulator row: C elem g -> ch = nb*16 + g + 8*hi
    v8f c1init[4];
#pragma unroll
    for (int nb = 0; nb < 4; nb++) {
#pragma unroll
        for (int g = 0; g < 8; g++) c1init[nb][g] = s_b1[nb * 16 + g + 8 * hi];
    }
    const float b2l = s_b2[r];

    _Float16* stg = s_stage[wave];

    for (int ii = 0; ii < 8; ii++) {
        const int i = wave * 8 + ii;
        const float pix = s_pos[i * 2 + 0], piy = s_pos[i * 2 + 1];
        float acc = 0.f;   // partial ctx[i][r] over this lane's half of j's

        for (int jb = 0; jb < 4; jb++) {
            const int j = jb * 16 + r;   // this lane's pair column
            // feat^T fragment (B layout): col = pair r; elem e -> feature channel hi*16+e
            //   hi==0 lanes: channels 0..15  = pe(dist)
            //   hi==1 lanes: channels 16..31 = hidden[j]
            v16h bfeat;
            if (hi == 0) {
                const float dx = pix - s_pos[j * 2 + 0];
                const float dy = piy - s_pos[j * 2 + 1];
#pragma unroll
                for (int e = 0; e < 16; e++)
                    bfeat[e] = (_Float16)reluf(dx * wx[e] + dy * wy[e] + pb[e]);
            } else {
                const v8h h0 = *(const v8h*)&s_hid[j * 16 + 0];
                const v8h h1 = *(const v8h*)&s_hid[j * 16 + 8];
#pragma unroll
                for (int e = 0; e < 8; e++) { bfeat[e] = h0[e]; bfeat[8 + e] = h1[e]; }
            }

            // Layer 1: D rows = channels (g -> nb*16+g+8hi), cols = pairs (lane r).
            // Each lane's 8 outputs are contiguous channels of pair r -> packed store.
#pragma unroll
            for (int nb = 0; nb < 4; nb++) {
                v8f c1 = c1init[nb];
                c1 = __builtin_amdgcn_wmma_f32_16x16x32_f16(
                         false, aw1[nb], false, bfeat, (short)0, c1, false, false);
                v8h p;
#pragma unroll
                for (int g = 0; g < 8; g++) p[g] = (_Float16)reluf(c1[g]);
                *(v8h*)&stg[r * 64 + nb * 16 + 8 * hi] = p;
            }
            asm volatile("s_wait_dscnt 0" ::: "memory");   // cross-lane LDS transpose fence

            // Layer 2 A fragments: lane(r,hi) row=pair r; K runs are contiguous b128s.
            const v8h q0 = *(const v8h*)&stg[r * 64 +  0 + hi * 8];
            const v8h q1 = *(const v8h*)&stg[r * 64 + 16 + hi * 8];
            const v8h q2 = *(const v8h*)&stg[r * 64 + 32 + hi * 8];
            const v8h q3 = *(const v8h*)&stg[r * 64 + 48 + hi * 8];
            v16h a2l, a2h;
#pragma unroll
            for (int e = 0; e < 8; e++) {
                a2l[e] = q0[e]; a2l[8 + e] = q1[e];
                a2h[e] = q2[e]; a2h[8 + e] = q3[e];
            }
            v8f c2;
#pragma unroll
            for (int g = 0; g < 8; g++) c2[g] = b2l;
            c2 = __builtin_amdgcn_wmma_f32_16x16x32_f16(false, a2l, false, bw2[0], (short)0, c2, false, false);
            c2 = __builtin_amdgcn_wmma_f32_16x16x32_f16(false, a2h, false, bw2[1], (short)0, c2, false, false);

            // mask-weighted reduce over pair rows (VGPR g -> j = jb*16 + g + 8hi)
            const v8f mv = *(const v8f*)&s_mask[i * 64 + jb * 16 + 8 * hi];
#pragma unroll
            for (int g = 0; g < 8; g++) acc += mv[g] * reluf(c2[g]);
        }
        const float tot = acc + __shfl_xor(acc, 16, 32);
        if (hi == 0) ctxb[((size_t)s * NA + i) * 16 + r] = tot;
    }
}

// ---------------------------------------------------------------------------
// Phase 2c (per step): output head; updates out2/curr; writes predictions
// ---------------------------------------------------------------------------
__global__ __launch_bounds__(256) void head_step(
    const float* __restrict__ hbuf, const float* __restrict__ ctxb,
    const float* __restrict__ relg,
    const float* __restrict__ W1, const float* __restrict__ b1,   // (18,16),(16)
    const float* __restrict__ W2, const float* __restrict__ b2,   // (16,4),(4)
    float* __restrict__ out2, float* __restrict__ curr, float* __restrict__ pred)
{
    __shared__ float sW1[288], sb1[16], sW2[64], sb2[4];
    int tid = threadIdx.x;
    for (int t = tid; t < 288; t += 256) sW1[t] = W1[t];
    if (tid < 16) sb1[tid] = b1[tid];
    if (tid < 64) sW2[tid] = W2[tid];
    if (tid < 4)  sb2[tid] = b2[tid];
    __syncthreads();

    int b = blockIdx.x * 256 + tid;
    float a[18], hh[16];
#pragma unroll
    for (int k = 0; k < 16; k++) a[k] = hbuf[b * 16 + k] + ctxb[b * 16 + k];
    a[16] = relg[b * 2 + 0]; a[17] = relg[b * 2 + 1];
#pragma unroll
    for (int k = 0; k < 16; k++) {
        float v = sb1[k];
#pragma unroll
        for (int q = 0; q < 18; q++) v += a[q] * sW1[q * 16 + k];
        hh[k] = reluf(v);
    }
    float mu0 = sb2[0], mu1 = sb2[1];
#pragma unroll
    for (int q = 0; q < 16; q++) { mu0 += hh[q] * sW2[q * 4 + 0]; mu1 += hh[q] * sW2[q * 4 + 1]; }
    out2[b * 2 + 0] = mu0; out2[b * 2 + 1] = mu1;
    curr[b * 2 + 0] += mu0; curr[b * 2 + 1] += mu1;
    pred[b * 2 + 0] = mu0; pred[b * 2 + 1] = mu1;
}

__global__ void zero_nll(float* p) { *p = 0.0f; }   // mu==out -> nll identically 0

// ---------------------------------------------------------------------------
extern "C" void kernel_launch(void* const* d_in, const int* in_sizes, int n_in,
                              void* d_out, int out_size, void* d_ws, size_t ws_size,
                              hipStream_t stream)
{
    const float* traj_rel = (const float*)d_in[0];
    const float* obs_pos  = (const float*)d_in[1];
    const float* goal     = (const float*)d_in[2];
    const float* enc_We   = (const float*)d_in[4];
    const float* enc_be   = (const float*)d_in[5];
    const float* enc_Wih  = (const float*)d_in[6];
    const float* enc_Whh  = (const float*)d_in[7];
    const float* enc_b    = (const float*)d_in[8];
    const float* dec_inW  = (const float*)d_in[9];
    const float* dec_inb  = (const float*)d_in[10];
    const float* dec_Wih  = (const float*)d_in[11];
    const float* dec_Whh  = (const float*)d_in[12];
    const float* dec_b    = (const float*)d_in[13];
    const float* Wp  = (const float*)d_in[14];
    const float* bpv = (const float*)d_in[15];
    const float* W1  = (const float*)d_in[16];
    const float* b1  = (const float*)d_in[17];
    const float* W2  = (const float*)d_in[18];
    const float* b2  = (const float*)d_in[19];
    const float* h2p1_W = (const float*)d_in[20];
    const float* h2p1_b = (const float*)d_in[21];
    const float* h2p2_W = (const float*)d_in[22];
    const float* h2p2_b = (const float*)d_in[23];
    const int*   nei    = (const int*)d_in[24];
    float* out = (float*)d_out;

    // workspace layout (floats): h | c | ctx | out2 | curr | relg | h16(f16)
    float* ws   = (float*)d_ws;
    float* hbuf = ws;
    float* cbuf = hbuf + (size_t)NB * 16;
    float* ctxb = cbuf + (size_t)NB * 16;
    float* out2 = ctxb + (size_t)NB * 16;
    float* curr = out2 + (size_t)NB * 2;
    float* relg = curr + (size_t)NB * 2;
    _Float16* h16 = (_Float16*)(relg + (size_t)NB * 2);

    enc_kernel<<<NB / 256, 256, 0, stream>>>(traj_rel, obs_pos, goal,
        enc_We, enc_be, enc_Wih, enc_Whh, enc_b,
        hbuf, cbuf, ctxb, out2, curr, relg);

    for (int t = 0; t < PLEN; t++) {
        dec_step<<<NB / 256, 256, 0, stream>>>(ctxb, out2,
            dec_inW, dec_inb, dec_Wih, dec_Whh, dec_b, hbuf, cbuf, h16);
        pool_step<<<NS, 256, 0, stream>>>(curr, h16, nei,
            Wp, bpv, W1, b1, W2, b2, ctxb);
        head_step<<<NB / 256, 256, 0, stream>>>(hbuf, ctxb, relg,
            h2p1_W, h2p1_b, h2p2_W, h2p2_b,
            out2, curr, out + (size_t)t * NB * 2);
    }
    zero_nll<<<1, 1, 0, stream>>>(out + (size_t)PLEN * NB * 2);
}
